// LocalPoolPointnet_4372276707771
// MI455X (gfx1250) — matrix operand order repaired and checked by reading, not compiled, not run on hardware
//
#include <hip/hip_runtime.h>
#include <hip/hip_bf16.h>

// ---------------------------------------------------------------------------
// LocalPoolPointnet on gfx1250 (CDNA5, wave32).
// GEMMs use v_wmma_f32_16x16x32_bf16 (bf16 A/B, f32 accumulate).
// M=32 tiling: each B fragment feeds two WMMAs (two 16-row tiles).
// ---------------------------------------------------------------------------

typedef __attribute__((ext_vector_type(16))) __bf16 v16bf;
typedef __attribute__((ext_vector_type(8)))  float  v8f;

using u16 = unsigned short;
using u32 = unsigned int;

#define NBATCH 2
#define NPTS   32768
#define BT     65536          // NBATCH * NPTS
#define HID    128
#define RESO   128
#define R2     16384          // RESO*RESO
#define NBLK   5
#define INVNORM (1.0f / 1.101f)   // 1/(1 + PAD + 1e-3)

// ---------------- small helpers ----------------

__device__ inline u16 f2bf(float f) {
    u32 u = __float_as_uint(f);
    u32 r = u + 0x7FFFu + ((u >> 16) & 1u);   // round-to-nearest-even
    return (u16)(r >> 16);
}
__device__ inline float bf2f(u16 h) { return __uint_as_float(((u32)h) << 16); }

__device__ inline u32 f32_to_ordered(float v) {
    u32 u = __float_as_uint(v);
    return (u & 0x80000000u) ? ~u : (u | 0x80000000u);
}
__device__ inline float ordered_to_f32(u32 k) {
    u32 u = (k & 0x80000000u) ? (k & 0x7FFFFFFFu) : ~k;
    return __uint_as_float(u);
}

// A-fragment (16-bit A 16x32 ISA layout), rows in LDS row-major (stride halves)
__device__ inline v16bf load_a_frag(const u16* s, int stride, int k0, int lane) {
    int m    = lane & 15;
    int koff = (lane < 16) ? 0 : 8;
    const u16* base = s + m * stride + k0 + koff;
    union { v16bf v; uint4 q[2]; } f;
    f.q[0] = *(const uint4*)(base);        // K  +0..+7
    f.q[1] = *(const uint4*)(base + 16);   // K +16..+23
    return f.v;
}

// B-fragment (16-bit B 32x16 ISA layout) from transposed weights wT[N][K] bf16
__device__ inline v16bf load_b_frag(const u16* wT, int K, int n, int k0, int lane) {
    int koff = (lane < 16) ? 0 : 8;
    const u16* base = wT + n * K + k0 + koff;
    union { v16bf v; uint4 q[2]; } f;
    f.q[0] = *(const uint4*)(base);        // K  +0..+7
    f.q[1] = *(const uint4*)(base + 16);   // K +16..+23
    return f.v;
}

__device__ inline v8f wmma_bf16(v16bf a, v16bf b, v8f c) {
    return __builtin_amdgcn_wmma_f32_16x16x32_bf16(
        false, a, false, b, (short)0, c, false, false);
}

// ---------------- utility kernels ----------------

__global__ void fill_u32_kernel(u32* dst, u32 val, int n) {
    int stride = gridDim.x * blockDim.x;
    for (int i = blockIdx.x * blockDim.x + threadIdx.x; i < n; i += stride)
        dst[i] = val;
}

// src (K,N) f32 row-major -> dst (N,K) bf16
__global__ void conv_transpose_kernel(const float* __restrict__ src,
                                      u16* __restrict__ dst, int K, int N) {
    int t = blockIdx.x * blockDim.x + threadIdx.x;
    if (t >= K * N) return;
    int k = t / N, n = t % N;
    dst[n * K + k] = f2bf(src[t]);
}

// ---------------- point prep: indices + fc_pos ----------------

__global__ void prep_kernel(const float* __restrict__ p,
                            const float* __restrict__ W,   // (3,256)
                            const float* __restrict__ b,   // (256)
                            u16* __restrict__ xbuf,        // (BT,256) bf16
                            int* __restrict__ idxbuf) {    // (3,BT)
    int pt = blockIdx.x;
    int o  = threadIdx.x;                  // 256 threads = output features
    float x0 = p[pt * 3 + 0];
    float x1 = p[pt * 3 + 1];
    float x2 = p[pt * 3 + 2];
    if (o < 3) {
        // plane dims: 0:(x0,x2) xz  1:(x0,x1) xy  2:(x1,x2) yz
        float u = (o == 0) ? x0 : (o == 1) ? x0 : x1;
        float v = (o == 0) ? x2 : (o == 1) ? x1 : x2;
        float nu = fminf(fmaxf(u * INVNORM + 0.5f, 0.0f), 1.0f - 1e-5f);
        float nv = fminf(fmaxf(v * INVNORM + 0.5f, 0.0f), 1.0f - 1e-5f);
        int gu = (int)(nu * RESO);
        int gv = (int)(nv * RESO);
        idxbuf[o * BT + pt] = gu + RESO * gv;
    }
    float acc = b[o] + x0 * W[o] + x1 * W[256 + o] + x2 * W[512 + o];
    xbuf[pt * 256 + o] = f2bf(acc);
}

// ---------------- ResNet block (WMMA, M=32 per workgroup) ----------------
// 256 threads = 8 wave32.  Each wave owns a 16-wide N slice and computes two
// 16x16 output tiles (rows 0-15 and 16-31) sharing every B fragment.

__global__ void __launch_bounds__(256)
resnet_block_kernel(const u16* __restrict__ xbuf,   // (BT,256) bf16 (in-place safe)
                    const u16* __restrict__ w0T,    // (128,256) bf16
                    const float* __restrict__ b0,
                    const u16* __restrict__ w1T,    // (128,128) bf16
                    const float* __restrict__ b1,
                    const u16* __restrict__ wsT,    // (128,256) bf16
                    u16* __restrict__ netOut,       // (BT,128) bf16
                    u16* __restrict__ xOut) {       // (BT,256) bf16, cols [0,128)
    __shared__ __align__(16) u16 sXraw[32 * 256];
    __shared__ __align__(16) u16 sXrel[32 * 256];
    __shared__ __align__(16) u16 sH[32 * 128];

    const int tid  = threadIdx.x;
    const int wave = tid >> 5;
    const int lane = tid & 31;
    const int pt0  = blockIdx.x * 32;
    const int n    = wave * 16 + (lane & 15);

    // Prefetch weight rows consumed late in the kernel (W1, Ws slices).
    if (lane < 16) {
        __builtin_prefetch(w1T + n * 128, 0, 1);
        __builtin_prefetch(w1T + n * 128 + 128, 0, 1);
        #pragma unroll
        for (int k0 = 0; k0 < 256; k0 += 64)
            __builtin_prefetch(wsT + n * 256 + k0, 0, 1);
    }

    // stage x tile (raw + relu) : 32 pts x 256 halves = 1024 uint4
    for (int i = tid; i < 1024; i += 256) {
        uint4 v = ((const uint4*)xbuf)[(size_t)pt0 * 32 + i];
        u32* w = (u32*)&v;
        ((uint4*)sXraw)[i] = v;
        uint4 r;
        u32* rw = (u32*)&r;
        #pragma unroll
        for (int j = 0; j < 4; ++j) {
            u32 a = w[j];
            u32 lo = (a & 0x8000u)     ? 0u : (a & 0xFFFFu);
            u32 hi = (a & 0x80000000u) ? 0u : (a & 0xFFFF0000u);
            rw[j] = lo | hi;
        }
        ((uint4*)sXrel)[i] = r;
    }
    __syncthreads();

    // GEMM1: h = relu( relu(x) @ W0 + b0 )   (K = 256)
    v8f acc0 = {}, acc1 = {};
    #pragma unroll
    for (int k0 = 0; k0 < 256; k0 += 32) {
        v16bf bfr = load_b_frag(w0T, 256, n, k0, lane);
        acc0 = wmma_bf16(load_a_frag(sXrel,            256, k0, lane), bfr, acc0);
        acc1 = wmma_bf16(load_a_frag(sXrel + 16 * 256, 256, k0, lane), bfr, acc1);
    }
    {
        float bias0 = b0[n];
        int mb = (lane >= 16) ? 8 : 0;
        #pragma unroll
        for (int r = 0; r < 8; ++r) {
            sH[(r + mb) * 128 + n]      = f2bf(fmaxf(acc0[r] + bias0, 0.0f));
            sH[(r + mb + 16) * 128 + n] = f2bf(fmaxf(acc1[r] + bias0, 0.0f));
        }
    }
    __syncthreads();

    // GEMM2: dx = h @ W1 + b1  (K = 128)
    v8f accD0 = {}, accD1 = {};
    #pragma unroll
    for (int k0 = 0; k0 < 128; k0 += 32) {
        v16bf bfr = load_b_frag(w1T, 128, n, k0, lane);
        accD0 = wmma_bf16(load_a_frag(sH,            128, k0, lane), bfr, accD0);
        accD1 = wmma_bf16(load_a_frag(sH + 16 * 128, 128, k0, lane), bfr, accD1);
    }

    // GEMM3: shortcut = x @ Ws  (K = 256)
    v8f accS0 = {}, accS1 = {};
    #pragma unroll
    for (int k0 = 0; k0 < 256; k0 += 32) {
        v16bf bfr = load_b_frag(wsT, 256, n, k0, lane);
        accS0 = wmma_bf16(load_a_frag(sXraw,            256, k0, lane), bfr, accS0);
        accS1 = wmma_bf16(load_a_frag(sXraw + 16 * 256, 256, k0, lane), bfr, accS1);
    }

    float bias1 = b1[n];
    int mb = (lane >= 16) ? 8 : 0;
    #pragma unroll
    for (int r = 0; r < 8; ++r) {
        int ptA = pt0 + r + mb;
        int ptB = ptA + 16;
        u16 oA = f2bf(accD0[r] + bias1 + accS0[r]);
        u16 oB = f2bf(accD1[r] + bias1 + accS1[r]);
        netOut[(size_t)ptA * 128 + n] = oA;
        xOut[(size_t)ptA * 256 + n]   = oA;
        netOut[(size_t)ptB * 128 + n] = oB;
        xOut[(size_t)ptB * 256 + n]   = oB;
    }
}

// ---------------- final projection c = net @ Wc + bc (WMMA, M=32) ----------------

__global__ void __launch_bounds__(256)
fcc_kernel(const u16* __restrict__ netbuf,   // (BT,128) bf16
           const u16* __restrict__ wcT,      // (128,128) bf16
           const float* __restrict__ bc,
           float* __restrict__ cbuf) {       // (BT,128) f32
    __shared__ __align__(16) u16 sN[32 * 128];
    const int tid  = threadIdx.x;
    const int wave = tid >> 5;
    const int lane = tid & 31;
    const int pt0  = blockIdx.x * 32;
    const int n    = wave * 16 + (lane & 15);

    // 32 pts x 128 halves = 512 uint4
    for (int i = tid; i < 512; i += 256)
        ((uint4*)sN)[i] = ((const uint4*)netbuf)[(size_t)pt0 * 16 + i];
    __syncthreads();

    v8f acc0 = {}, acc1 = {};
    #pragma unroll
    for (int k0 = 0; k0 < 128; k0 += 32) {
        v16bf bfr = load_b_frag(wcT, 128, n, k0, lane);
        acc0 = wmma_bf16(load_a_frag(sN,            128, k0, lane), bfr, acc0);
        acc1 = wmma_bf16(load_a_frag(sN + 16 * 128, 128, k0, lane), bfr, acc1);
    }

    float bias = bc[n];
    int mb = (lane >= 16) ? 8 : 0;
    #pragma unroll
    for (int r = 0; r < 8; ++r) {
        cbuf[(size_t)(pt0 + r + mb) * 128 + n]      = acc0[r] + bias;
        cbuf[(size_t)(pt0 + r + mb + 16) * 128 + n] = acc1[r] + bias;
    }
}

// ---------------- scatter-max / gather ----------------

__global__ void scatter_max_kernel(const u16* __restrict__ netbuf,
                                   const int* __restrict__ idxbuf,
                                   u32* __restrict__ pool) {  // (3,B,R2,128)
    int blk   = blockIdx.x;          // plane*BT + pt
    int plane = blk / BT;
    int pt    = blk % BT;
    int b     = pt / NPTS;
    int cell  = idxbuf[plane * BT + pt];
    int chan  = threadIdx.x;
    float v = bf2f(netbuf[(size_t)pt * 128 + chan]);
    u32 key = f32_to_ordered(v);
    atomicMax(&pool[(((size_t)(plane * NBATCH + b) * R2 + cell) * 128) + chan], key);
}

__global__ void gather_kernel(const u32* __restrict__ pool,
                              const int* __restrict__ idxbuf,
                              u16* __restrict__ xbuf) {       // cols [128,256)
    int pt   = blockIdx.x;
    int b    = pt / NPTS;
    int chan = threadIdx.x;
    float s = 0.0f;
    #pragma unroll
    for (int plane = 0; plane < 3; ++plane) {
        int cell = idxbuf[plane * BT + pt];
        u32 key = pool[(((size_t)(plane * NBATCH + b) * R2 + cell) * 128) + chan];
        s += ordered_to_f32(key);
    }
    xbuf[(size_t)pt * 256 + 128 + chan] = f2bf(s);
}

// ---------------- scatter-mean into output ----------------

__global__ void scatter_add_kernel(const float* __restrict__ cbuf,
                                   const int* __restrict__ idxbuf,
                                   float* __restrict__ out,     // (3,B,128,R2)
                                   float* __restrict__ counts) {// (3,B,R2)
    int blk   = blockIdx.x;
    int plane = blk / BT;
    int pt    = blk % BT;
    int b     = pt / NPTS;
    int cell  = idxbuf[plane * BT + pt];
    int chan  = threadIdx.x;
    float v = cbuf[(size_t)pt * 128 + chan];
    atomicAdd(&out[((size_t)(plane * NBATCH + b) * 128 + chan) * R2 + cell], v);
    if (chan == 0)
        atomicAdd(&counts[(size_t)(plane * NBATCH + b) * R2 + cell], 1.0f);
}

__global__ void divide_kernel(float* __restrict__ out,
                              const float* __restrict__ counts, int n) {
    int stride = gridDim.x * blockDim.x;
    for (int e = blockIdx.x * blockDim.x + threadIdx.x; e < n; e += stride) {
        int pb   = e / (128 * R2);
        int cell = e % R2;
        float c = counts[(size_t)pb * R2 + cell];
        out[e] = out[e] / fmaxf(c, 1.0f);
    }
}

// ---------------- host launcher ----------------

extern "C" void kernel_launch(void* const* d_in, const int* in_sizes, int n_in,
                              void* d_out, int out_size, void* d_ws, size_t ws_size,
                              hipStream_t stream) {
    const float* p        = (const float*)d_in[0];
    const float* fc_pos_W = (const float*)d_in[1];
    const float* fc_pos_b = (const float*)d_in[2];
    const float* W0       = (const float*)d_in[3];   // (5,256,128)
    const float* b0       = (const float*)d_in[4];   // (5,128)
    const float* W1       = (const float*)d_in[5];   // (5,128,128)
    const float* b1       = (const float*)d_in[6];   // (5,128)
    const float* Ws       = (const float*)d_in[7];   // (5,256,128)
    const float* Wc       = (const float*)d_in[8];   // (128,128)
    const float* bc       = (const float*)d_in[9];   // (128)
    float* out = (float*)d_out;

    // workspace layout (bytes)
    char* ws = (char*)d_ws;
    const size_t OFF_X    = 0;                                // xbuf (BT,256) bf16
    const size_t OFF_NET  = OFF_X   + (size_t)BT * 256 * 2;   // netbuf (BT,128) bf16
    const size_t OFF_IDX  = OFF_NET + (size_t)BT * 128 * 2;   // idx (3,BT) i32
    const size_t OFF_POOL = OFF_IDX + (size_t)3 * BT * 4;     // pool keys / cbuf (shared)
    const size_t OFF_CNT  = OFF_POOL + (size_t)3 * NBATCH * R2 * 128 * 4;
    const size_t OFF_W    = OFF_CNT  + (size_t)3 * NBATCH * R2 * 4;

    u16*   xbuf   = (u16*)(ws + OFF_X);
    u16*   netbuf = (u16*)(ws + OFF_NET);
    int*   idxbuf = (int*)(ws + OFF_IDX);
    u32*   pool   = (u32*)(ws + OFF_POOL);
    float* cbuf   = (float*)(ws + OFF_POOL);   // reuse: needed only after the loop
    float* counts = (float*)(ws + OFF_CNT);
    u16*   wbuf   = (u16*)(ws + OFF_W);

    // per-block bf16 transposed weights: [W0T 256x128][W1T][WsT] per block, then fc_cT
    const int W0T_HALVES = 256 * 128, W1T_HALVES = 128 * 128, WST_HALVES = 256 * 128;
    const int BLK_HALVES = W0T_HALVES + W1T_HALVES + WST_HALVES;   // 81920
    u16* fcT = wbuf + (size_t)NBLK * BLK_HALVES;

    for (int i = 0; i < NBLK; ++i) {
        conv_transpose_kernel<<<(256 * 128 + 255) / 256, 256, 0, stream>>>(
            W0 + (size_t)i * 256 * 128, wbuf + (size_t)i * BLK_HALVES, 256, 128);
        conv_transpose_kernel<<<(128 * 128 + 255) / 256, 256, 0, stream>>>(
            W1 + (size_t)i * 128 * 128, wbuf + (size_t)i * BLK_HALVES + W0T_HALVES, 128, 128);
        conv_transpose_kernel<<<(256 * 128 + 255) / 256, 256, 0, stream>>>(
            Ws + (size_t)i * 256 * 128,
            wbuf + (size_t)i * BLK_HALVES + W0T_HALVES + W1T_HALVES, 256, 128);
    }
    conv_transpose_kernel<<<(128 * 128 + 255) / 256, 256, 0, stream>>>(Wc, fcT, 128, 128);

    // indices + fc_pos -> xbuf
    prep_kernel<<<BT, 256, 0, stream>>>(p, fc_pos_W, fc_pos_b, xbuf, idxbuf);

    const int POOL_N = 3 * NBATCH * R2 * 128;       // 12,582,912
    for (int i = 0; i < NBLK; ++i) {
        if (i > 0) {
            fill_u32_kernel<<<4096, 256, 0, stream>>>(pool, 0u, POOL_N);
            scatter_max_kernel<<<3 * BT, 128, 0, stream>>>(netbuf, idxbuf, pool);
            gather_kernel<<<BT, 128, 0, stream>>>(pool, idxbuf, xbuf);
        }
        const u16* w0T = wbuf + (size_t)i * BLK_HALVES;
        const u16* w1T = w0T + W0T_HALVES;
        const u16* wsT = w1T + W1T_HALVES;
        resnet_block_kernel<<<BT / 32, 256, 0, stream>>>(
            xbuf, w0T, b0 + (size_t)i * 128, w1T, b1 + (size_t)i * 128, wsT,
            netbuf, xbuf);
    }

    // c = net @ Wc + bc   (cbuf aliases pool region; pool no longer needed)
    fcc_kernel<<<BT / 32, 256, 0, stream>>>(netbuf, fcT, bc, cbuf);

    // scatter-mean into output
    fill_u32_kernel<<<4096, 256, 0, stream>>>((u32*)out, 0u, out_size);
    fill_u32_kernel<<<96, 256, 0, stream>>>((u32*)counts, 0u, 3 * NBATCH * R2);
    scatter_add_kernel<<<3 * BT, 128, 0, stream>>>(cbuf, idxbuf, out, counts);
    divide_kernel<<<4096, 256, 0, stream>>>(out, counts, out_size);
}
